// HetGATPyG_55602646614360
// MI455X (gfx1250) — matrix operand order repaired and checked by reading, not compiled, not run on hardware
//
#include <hip/hip_runtime.h>
#include <hip/hip_bf16.h>
#include <stdint.h>

// ---------------- CDNA5 WMMA types ----------------
typedef __attribute__((ext_vector_type(16))) __bf16 v16bf;
typedef __attribute__((ext_vector_type(8)))  float  v8f;
typedef __attribute__((ext_vector_type(4)))  unsigned int u32x4;

#define NN   50000
#define FEATN 64
#define DD   128
#define HH   4
#define CC   32
#define FFN_ 512
#define EE   600000
#define TOTE (EE + NN)

__device__ __forceinline__ unsigned short f2bf(float f) {
  unsigned b = __float_as_uint(f);
  unsigned r = (b + 0x7FFFu + ((b >> 16) & 1u)) >> 16;   // RNE
  return (unsigned short)r;
}
// monotone float<->uint map for atomic max on floats
__device__ __forceinline__ unsigned encf(float f) {
  unsigned b = __float_as_uint(f);
  return (b & 0x80000000u) ? ~b : (b | 0x80000000u);
}
__device__ __forceinline__ float decf(unsigned u) {
  unsigned b = (u & 0x80000000u) ? (u ^ 0x80000000u) : ~u;
  return __uint_as_float(b);
}

union FragAB { v16bf v; u32x4 q[2]; };

// ---------------- utility kernels ----------------
__global__ void fill_u32_k(unsigned* p, unsigned v, int n) {
  int i = blockIdx.x * 256 + threadIdx.x;
  if (i < n) p[i] = v;
}

__global__ void f32_to_bf16_k(const float* __restrict__ s, unsigned short* __restrict__ d, int n) {
  int i = blockIdx.x * 256 + threadIdx.x;
  if (i < n) d[i] = f2bf(s[i]);
}

// Pack fp32 weight [K x Nout] (row-major) into exact WMMA B-fragment order, bf16.
// Fragment (kt, ct) occupies 512 bf16: lane*16 + e, where lane = n + 16g, K = kt*32 + (e < 8 ? 8g + e : 16 + 8g + (e - 8)).
__global__ void pack_weight_k(const float* __restrict__ W, unsigned short* __restrict__ dst,
                              int K, int Nout) {
  int idx = blockIdx.x * 256 + threadIdx.x;
  int total = K * Nout;
  if (idx >= total) return;
  int ntiles = Nout >> 4;
  int frag = idx >> 9, r = idx & 511, lane = r >> 4, e = r & 15;
  int kt = frag / ntiles, ct = frag - kt * ntiles;
  int n = lane & 15, g = lane >> 4;
  int kk = kt * 32 + (e < 8 ? 8 * g + e : 16 + 8 * g + (e - 8));
  dst[idx] = f2bf(W[(size_t)kk * Nout + ct * 16 + n]);
}

// ---------------- WMMA GEMM: OUT[M x Nout] = act(Abf16[M x K] @ Wpacked + bias) ----------------
// grid.x = M/16 row tiles, 256 thr = 8 waves; wave w covers column tiles w, w+8, ...
__global__ __launch_bounds__(256)
void gemm_bf16_k(const unsigned short* __restrict__ A, const unsigned short* __restrict__ Wp,
                 const float* __restrict__ bias, float* __restrict__ out,
                 unsigned short* __restrict__ outbf, int K, int Nout, int relu) {
  const int wave = threadIdx.x >> 5, lane = threadIdx.x & 31;
  const int ntiles = Nout >> 4, ktiles = K >> 5;
  const int mn = lane & 15, g = lane >> 4;
  const int row = blockIdx.x * 16 + mn;
  const unsigned short* arow = A + (size_t)row * K;
  const v8f vzero = {0.f,0.f,0.f,0.f,0.f,0.f,0.f,0.f};

  for (int ct = wave; ct < ntiles; ct += 8) {
    v8f acc = vzero;
    for (int kt = 0; kt < ktiles; ++kt) {
      FragAB a, b;
      const unsigned short* ap = arow + kt * 32 + 8 * g;   // K chunks of 8 are contiguous
      a.q[0] = *(const u32x4*)ap;          // K = kt*32 + 8g .. +7
      a.q[1] = *(const u32x4*)(ap + 16);   // K = kt*32 + 16 + 8g .. +7
      const u32x4* bp = (const u32x4*)(Wp + ((size_t)(kt * ntiles + ct) << 9) + lane * 16);
      b.q[0] = bp[0]; b.q[1] = bp[1];
      acc = __builtin_amdgcn_wmma_f32_16x16x32_bf16(false, a.v, false, b.v,
                                                    (short)0, acc, false, false);
    }
    const int col = ct * 16 + mn;
    const float bv = bias ? bias[col] : 0.f;
#pragma unroll
    for (int r = 0; r < 8; ++r) {
      int m = blockIdx.x * 16 + r + 8 * g;      // D layout: lane holds (m = r+8g, n = lane&15)
      float v = acc[r] + bv;
      if (relu) v = fmaxf(v, 0.f);
      if (out)   out  [(size_t)m * Nout + col] = v;
      if (outbf) outbf[(size_t)m * Nout + col] = f2bf(v);
    }
  }
}

// ---------------- LayerNorm (one wave per node, D=128 => 4 ch/lane) ----------------
__global__ __launch_bounds__(256)
void ln_k(const float* __restrict__ xin, const float* __restrict__ delta,
          const float* __restrict__ bias, const float* __restrict__ gamma,
          const float* __restrict__ beta, float* __restrict__ xout,
          unsigned short* __restrict__ bfout) {
  int node = blockIdx.x * 8 + (threadIdx.x >> 5);
  int lane = threadIdx.x & 31;
  if (node >= NN) return;
  size_t base = (size_t)node * DD;
  float v[4], s = 0.f, sq = 0.f;
#pragma unroll
  for (int j = 0; j < 4; ++j) {
    int c = lane + 32 * j;
    float t = xin[base + c];
    if (delta) t += delta[base + c];
    if (bias)  t += bias[c];
    v[j] = t; s += t; sq += t * t;
  }
#pragma unroll
  for (int o = 16; o >= 1; o >>= 1) { s += __shfl_xor(s, o, 32); sq += __shfl_xor(sq, o, 32); }
  float m = s * (1.f / DD);
  float var = sq * (1.f / DD) - m * m;
  float inv = rsqrtf(var + 1e-5f);
#pragma unroll
  for (int j = 0; j < 4; ++j) {
    int c = lane + 32 * j;
    float y = (v[j] - m) * inv * gamma[c] + beta[c];
    xout[base + c] = y;
    if (bfout) bfout[base + c] = f2bf(y);
  }
}

// ---------------- attention logits: per (node, head) dot over C=32 ----------------
__global__ void att_logits_k(const float* __restrict__ hW, const float* __restrict__ a_s,
                             const float* __restrict__ a_d, float* __restrict__ als,
                             float* __restrict__ ald) {
  int idx = blockIdx.x * 256 + threadIdx.x;
  if (idx >= NN * HH) return;
  int n = idx >> 2, h = idx & 3;
  const float* r = hW + (size_t)n * DD + h * CC;
  float s = 0.f, d = 0.f;
#pragma unroll
  for (int c = 0; c < CC; ++c) { float v = r[c]; s += v * a_s[h * CC + c]; d += v * a_d[h * CC + c]; }
  als[idx] = s; ald[idx] = d;
}

// ---------------- edge softmax pass 1: leaky-relu logits + segment max ----------------
__global__ void edge_max_k(const int* __restrict__ es, const int* __restrict__ ed,
                           const float* __restrict__ als, const float* __restrict__ ald,
                           float* __restrict__ alpha, unsigned* __restrict__ amax) {
  int i = blockIdx.x * 256 + threadIdx.x;
  if (i >= TOTE) return;
  int s = (i < EE) ? es[i] : (i - EE);
  int d = (i < EE) ? ed[i] : (i - EE);
#pragma unroll
  for (int h = 0; h < HH; ++h) {
    float a = als[s * HH + h] + ald[d * HH + h];
    a = a > 0.f ? a : 0.2f * a;                       // leaky_relu 0.2
    alpha[(size_t)i * HH + h] = a;
    atomicMax(&amax[d * HH + h], encf(a));
  }
}

// ---------------- pass 2: exp(alpha - max) + segment sum ----------------
__global__ void edge_expsum_k(const int* __restrict__ es, const int* __restrict__ ed,
                              float* __restrict__ alpha, const unsigned* __restrict__ amax,
                              float* __restrict__ denom) {
  int i = blockIdx.x * 256 + threadIdx.x;
  if (i >= TOTE) return;
  int d = (i < EE) ? ed[i] : (i - EE);
#pragma unroll
  for (int h = 0; h < HH; ++h) {
    float ex = __expf(alpha[(size_t)i * HH + h] - decf(amax[d * HH + h]));
    alpha[(size_t)i * HH + h] = ex;
    atomicAdd(&denom[d * HH + h], ex);
  }
}

// ---------------- pass 3: weighted message scatter (one wave per edge) ----------------
__global__ __launch_bounds__(256)
void edge_scatter_k(const int* __restrict__ es, const int* __restrict__ ed,
                    const float* __restrict__ alpha, const float* __restrict__ denom,
                    const float* __restrict__ hW, float* __restrict__ outacc) {
  int wv = blockIdx.x * 8 + (threadIdx.x >> 5);
  if (wv >= TOTE) return;
  int lane = threadIdx.x & 31;
  int s = (wv < EE) ? es[wv] : (wv - EE);
  int d = (wv < EE) ? ed[wv] : (wv - EE);
#pragma unroll
  for (int j = 0; j < 4; ++j) {                        // head j covers channels j*32..j*32+31
    float w = alpha[(size_t)wv * HH + j] / denom[d * HH + j];
    int c = j * 32 + lane;
    atomicAdd(&outacc[(size_t)d * DD + c], hW[(size_t)s * DD + c] * w);
  }
}

// ---------------- fused edge predictor: relu([h_s||h_d]@We1+be1)@We2+be2, relu ----------------
// one wave per 16 edges; K=256 gathered from src/dst bf16 rows; N=256 in two groups of 8 tiles.
__global__ __launch_bounds__(256)
void edge_predict_k(const unsigned short* __restrict__ hbf, const int* __restrict__ es,
                    const int* __restrict__ ed, const unsigned short* __restrict__ We1p,
                    const float* __restrict__ be1, const float* __restrict__ We2,
                    const float* __restrict__ be2, float* __restrict__ out) {
  int wv = blockIdx.x * 8 + (threadIdx.x >> 5);
  int lane = threadIdx.x & 31;
  if (wv >= EE / 16) return;
  const int mn = lane & 15, g = lane >> 4;
  const int e0 = wv * 16;
  const int se = es[e0 + mn], de = ed[e0 + mn];
  const unsigned short* rs = hbf + (size_t)se * DD;
  const unsigned short* rd = hbf + (size_t)de * DD;
  const v8f vzero = {0.f,0.f,0.f,0.f,0.f,0.f,0.f,0.f};

  float flow[8];
#pragma unroll
  for (int r = 0; r < 8; ++r) flow[r] = 0.f;

  for (int ng = 0; ng < 2; ++ng) {
    v8f acc[8];
#pragma unroll
    for (int t = 0; t < 8; ++t) acc[t] = vzero;
    for (int kt = 0; kt < 8; ++kt) {
      FragAB a;
      const unsigned short* rp = (kt < 4 ? rs + kt * 32 : rd + (kt - 4) * 32) + 8 * g;
      a.q[0] = *(const u32x4*)rp;
      a.q[1] = *(const u32x4*)(rp + 16);
#pragma unroll
      for (int t = 0; t < 8; ++t) {
        int ct = ng * 8 + t;
        const u32x4* bp = (const u32x4*)(We1p + ((size_t)(kt * 16 + ct) << 9) + lane * 16);
        FragAB b; b.q[0] = bp[0]; b.q[1] = bp[1];
        acc[t] = __builtin_amdgcn_wmma_f32_16x16x32_bf16(false, a.v, false, b.v,
                                                         (short)0, acc[t], false, false);
      }
    }
#pragma unroll
    for (int t = 0; t < 8; ++t) {
      int ct = ng * 8 + t;
      float b1v = be1[ct * 16 + mn], w2 = We2[ct * 16 + mn];
#pragma unroll
      for (int r = 0; r < 8; ++r) {
        float p = fmaxf(acc[t][r] + b1v, 0.f) * w2;   // this lane's n = mn, edge m = r+8g
        p += __shfl_xor(p, 1, 32); p += __shfl_xor(p, 2, 32);
        p += __shfl_xor(p, 4, 32); p += __shfl_xor(p, 8, 32);   // reduce across 16 n-lanes
        flow[r] += p;
      }
    }
  }
  if (mn == 0) {
    float bb = be2[0];
#pragma unroll
    for (int r = 0; r < 8; ++r)
      out[e0 + r + 8 * g] = fmaxf(flow[r] + bb, 0.f);
  }
}

// ---------------- host orchestration ----------------
extern "C" void kernel_launch(void* const* d_in, const int* in_sizes, int n_in,
                              void* d_out, int out_size, void* d_ws, size_t ws_size,
                              hipStream_t stream) {
  (void)in_sizes; (void)n_in; (void)out_size; (void)ws_size;
  const float* x    = (const float*)d_in[0];
  const int*   ei   = (const int*)d_in[1];
  const int*   vei  = (const int*)d_in[2];
  const float* Wp   = (const float*)d_in[3];
  const float* bp   = (const float*)d_in[4];
  const float* lnpg = (const float*)d_in[5];
  const float* lnpb = (const float*)d_in[6];
  const float* pW[2]   = {(const float*)d_in[7],  (const float*)d_in[19]};
  const float* pAS[2]  = {(const float*)d_in[8],  (const float*)d_in[20]};
  const float* pAD[2]  = {(const float*)d_in[9],  (const float*)d_in[21]};
  const float* pB[2]   = {(const float*)d_in[10], (const float*)d_in[22]};
  const float* pW1[2]  = {(const float*)d_in[11], (const float*)d_in[23]};
  const float* pB1[2]  = {(const float*)d_in[12], (const float*)d_in[24]};
  const float* pW2[2]  = {(const float*)d_in[13], (const float*)d_in[25]};
  const float* pB2[2]  = {(const float*)d_in[14], (const float*)d_in[26]};
  const float* pL1G[2] = {(const float*)d_in[15], (const float*)d_in[27]};
  const float* pL1B[2] = {(const float*)d_in[16], (const float*)d_in[28]};
  const float* pL2G[2] = {(const float*)d_in[17], (const float*)d_in[29]};
  const float* pL2B[2] = {(const float*)d_in[18], (const float*)d_in[30]};
  const float* We1 = (const float*)d_in[31];
  const float* be1 = (const float*)d_in[32];
  const float* We2 = (const float*)d_in[33];
  const float* be2 = (const float*)d_in[34];

  // workspace carve
  char* base = (char*)d_ws;
  size_t off = 0;
  auto carve = [&](size_t bytes) -> void* {
    void* p = base + off; off += (bytes + 255) & ~(size_t)255; return p;
  };
  float* hA  = (float*)carve((size_t)NN * DD * 4);      // current activations (fp32)
  float* hB  = (float*)carve((size_t)NN * DD * 4);      // gat-out accumulator / ffn out
  float* hW  = (float*)carve((size_t)NN * DD * 4);      // h @ W (fp32, needed for messages)
  unsigned short* hbf  = (unsigned short*)carve((size_t)NN * DD * 2);
  unsigned short* xbf  = (unsigned short*)carve((size_t)NN * FEATN * 2);
  unsigned short* f1bf = (unsigned short*)carve((size_t)NN * FFN_ * 2);
  float*    als   = (float*)carve((size_t)NN * HH * 4);
  float*    ald   = (float*)carve((size_t)NN * HH * 4);
  unsigned* amax  = (unsigned*)carve((size_t)NN * HH * 4);
  float*    denom = (float*)carve((size_t)NN * HH * 4);
  float*    alpha = (float*)carve((size_t)TOTE * HH * 4);
  unsigned short* packWp = (unsigned short*)carve((size_t)FEATN * DD * 2);
  unsigned short* packW[4], *packW1[4], *packW2[4];
  for (int l = 0; l < 4; ++l) packW[l]  = (unsigned short*)carve((size_t)DD * DD * 2);
  for (int l = 0; l < 4; ++l) packW1[l] = (unsigned short*)carve((size_t)DD * FFN_ * 2);
  for (int l = 0; l < 4; ++l) packW2[l] = (unsigned short*)carve((size_t)FFN_ * DD * 2);
  unsigned short* packWe1 = (unsigned short*)carve((size_t)256 * 256 * 2);

  dim3 B(256);
  // ---- pack all weights to bf16 WMMA B-fragment layout (tiny; once per launch) ----
  pack_weight_k<<<(FEATN * DD + 255) / 256, B, 0, stream>>>(Wp, packWp, FEATN, DD);
  for (int l = 0; l < 4; ++l) {
    int t = l >> 1, s = l & 1;
    pack_weight_k<<<(DD * DD + 255) / 256, B, 0, stream>>>(pW[t]  + (size_t)s * DD * DD,  packW[l],  DD,  DD);
    pack_weight_k<<<(DD * FFN_ + 255) / 256, B, 0, stream>>>(pW1[t] + (size_t)s * DD * FFN_, packW1[l], DD,  FFN_);
    pack_weight_k<<<(FFN_ * DD + 255) / 256, B, 0, stream>>>(pW2[t] + (size_t)s * FFN_ * DD, packW2[l], FFN_, DD);
  }
  pack_weight_k<<<(256 * 256 + 255) / 256, B, 0, stream>>>(We1, packWe1, 256, 256);

  // ---- preprocess: h = LN(relu(x @ Wp + bp)) ----
  f32_to_bf16_k<<<(NN * FEATN + 255) / 256, B, 0, stream>>>(x, xbf, NN * FEATN);
  gemm_bf16_k<<<NN / 16, B, 0, stream>>>(xbf, packWp, bp, hB, nullptr, FEATN, DD, 1);
  ln_k<<<NN / 8, B, 0, stream>>>(hB, nullptr, nullptr, lnpg, lnpb, hA, hbf);

  // ---- 2 virtual-link + 2 real-link encoder layers ----
  for (int l = 0; l < 4; ++l) {
    int t = l >> 1, s = l & 1;
    const int* esrc = (l < 2 ? vei : ei);
    const int* edst = esrc + EE;

    gemm_bf16_k<<<NN / 16, B, 0, stream>>>(hbf, packW[l], nullptr, hW, nullptr, DD, DD, 0);
    fill_u32_k<<<(NN * HH + 255) / 256, B, 0, stream>>>(amax, 0u, NN * HH);
    fill_u32_k<<<(NN * HH + 255) / 256, B, 0, stream>>>((unsigned*)denom, 0u, NN * HH);
    fill_u32_k<<<(NN * DD + 255) / 256, B, 0, stream>>>((unsigned*)hB, 0u, NN * DD);
    att_logits_k<<<(NN * HH + 255) / 256, B, 0, stream>>>(hW, pAS[t] + (size_t)s * HH * CC,
                                                          pAD[t] + (size_t)s * HH * CC, als, ald);
    edge_max_k   <<<(TOTE + 255) / 256, B, 0, stream>>>(esrc, edst, als, ald, alpha, amax);
    edge_expsum_k<<<(TOTE + 255) / 256, B, 0, stream>>>(esrc, edst, alpha, amax, denom);
    edge_scatter_k<<<(TOTE + 7) / 8, B, 0, stream>>>(esrc, edst, alpha, denom, hW, hB);
    ln_k<<<NN / 8, B, 0, stream>>>(hA, hB, pB[t] + (size_t)s * DD,
                                   pL1G[t] + (size_t)s * DD, pL1B[t] + (size_t)s * DD, hA, hbf);
    gemm_bf16_k<<<NN / 16, B, 0, stream>>>(hbf, packW1[l], pB1[t] + (size_t)s * FFN_,
                                           nullptr, f1bf, DD, FFN_, 1);
    gemm_bf16_k<<<NN / 16, B, 0, stream>>>(f1bf, packW2[l], pB2[t] + (size_t)s * DD,
                                           hB, nullptr, FFN_, DD, 0);
    ln_k<<<NN / 8, B, 0, stream>>>(hA, hB, nullptr,
                                   pL2G[t] + (size_t)s * DD, pL2B[t] + (size_t)s * DD, hA, hbf);
  }

  // ---- fused edge predictor ----
  edge_predict_k<<<(EE / 16 + 7) / 8, B, 0, stream>>>(hbf, ei, ei + EE, packWe1, be1, We2, be2,
                                                      (float*)d_out);
}
